// MutilHeadAttention_29429115912971
// MI455X (gfx1250) — matrix operand order repaired and checked by reading, not compiled, not run on hardware
//
#include <hip/hip_runtime.h>
#include <hip/hip_bf16.h>

typedef __attribute__((ext_vector_type(16))) __bf16         v16bf;
typedef __attribute__((ext_vector_type(8)))  float          v8f;
typedef __attribute__((ext_vector_type(4)))  float          v4f;
typedef __attribute__((ext_vector_type(16))) unsigned short v16u;
typedef __attribute__((ext_vector_type(8)))  unsigned short v8u;
typedef __attribute__((ext_vector_type(4)))  unsigned short v4u;

#define WMMA_BF16(a, b, c) \
  __builtin_amdgcn_wmma_f32_16x16x32_bf16(false, (a), false, (b), (short)0, (c), false, false)

// f32 -> bf16 via native fptrunc (RTNE); selects the HW convert on gfx1250
__device__ __forceinline__ unsigned short f2bf(float f) {
  __bf16 h = (__bf16)f;
  return __builtin_bit_cast(unsigned short, h);
}

// A-fragment (16x32 bf16, MxK) from row-major [row][k] bf16 buffer.
// ISA layout: lanes 0-15 M=0..15 (K 0..7 & 16..23), lanes 16-31 same M (K 8..15 & 24..31).
__device__ __forceinline__ v16bf frag_a(const unsigned short* p, int stride, int lane) {
  const int row = lane & 15, hs = lane >> 4;
  const unsigned short* r = p + row * stride + hs * 8;
  v8u lo = *(const v8u*)(r);        // K = hs*8 + 0..7
  v8u hi = *(const v8u*)(r + 16);   // K = 16 + hs*8 + 0..7
  v16u t;
#pragma unroll
  for (int i = 0; i < 8; ++i) { t[i] = lo[i]; t[i + 8] = hi[i]; }
  return __builtin_bit_cast(v16bf, t);
}

// B-fragment (32x16 bf16, KxN) from row-major [n][k] bf16 buffer (i.e. B^T storage).
// ISA layout: lanes 0-15 hold K 0..7 & 16..23 of column n=lane, lanes 16-31 hold K 8..15 & 24..31.
__device__ __forceinline__ v16bf frag_b(const unsigned short* p, int stride, int lane) {
  const int n = lane & 15, hs = lane >> 4;
  const unsigned short* r = p + n * stride + hs * 8;
  v8u lo = *(const v8u*)(r);
  v8u hi = *(const v8u*)(r + 16);
  v16u t;
#pragma unroll
  for (int i = 0; i < 8; ++i) { t[i] = lo[i]; t[i + 8] = hi[i]; }
  return __builtin_bit_cast(v16bf, t);
}

// ---------------------------------------------------------------------------
// Kernel 1: fused QKV projection. C[4096x1024] = X @ W^T, fp32 in -> bf16 out,
// written head-split: qh/kh = [64][1024][64], vT = [64][64][1024] (transposed).
// K-tile = 64 (two WMMA k-steps per LDS stage -> 8 WMMAs per barrier pair).
// ---------------------------------------------------------------------------
__global__ __launch_bounds__(128) void qkv_proj_kernel(
    const float* __restrict__ Q, const float* __restrict__ K, const float* __restrict__ V,
    const float* __restrict__ Wq, const float* __restrict__ Wk, const float* __restrict__ Wv,
    unsigned short* __restrict__ qh, unsigned short* __restrict__ kh,
    unsigned short* __restrict__ vT) {
  constexpr int LDA = 72;  // padded stride (ushorts): conflict-free, 16B-aligned rows
  __shared__ __align__(16) unsigned short As[64 * LDA];
  __shared__ __align__(16) unsigned short Bs[64 * LDA];

  const int sel = blockIdx.z;
  const float* X = sel == 0 ? Q : (sel == 1 ? K : V);
  const float* W = sel == 0 ? Wq : (sel == 1 ? Wk : Wv);
  const int m0 = blockIdx.y * 64;
  const int n0 = blockIdx.x * 64;
  const int t = threadIdx.x, lane = t & 31, w = t >> 5;
  const int wm = (w & 1) * 32, wn = (w >> 1) * 32;

  v8f acc00 = {}, acc01 = {}, acc10 = {}, acc11 = {};

  for (int kt = 0; kt < 1024; kt += 64) {
    if (kt + 64 < 1024) {  // cover next K-tile while this stage computes
      __builtin_prefetch(X + (size_t)(m0 + (t >> 1)) * 1024 + (kt + 64) + (t & 1) * 32, 0, 0);
      __builtin_prefetch(W + (size_t)(n0 + (t >> 1)) * 1024 + (kt + 64) + (t & 1) * 32, 0, 0);
    }
#pragma unroll
    for (int i = 0; i < 8; ++i) {  // 1024 float4 chunks over 128 threads
      const int c = i * 128 + t;
      const int row = c >> 4;          // 16 chunks per 64-wide row
      const int col = (c & 15) * 4;
      v4f xa = *(const v4f*)(X + (size_t)(m0 + row) * 1024 + kt + col);
      v4f wb = *(const v4f*)(W + (size_t)(n0 + row) * 1024 + kt + col);
      v4u pa = {f2bf(xa[0]), f2bf(xa[1]), f2bf(xa[2]), f2bf(xa[3])};
      v4u pb = {f2bf(wb[0]), f2bf(wb[1]), f2bf(wb[2]), f2bf(wb[3])};
      *(v4u*)(&As[row * LDA + col]) = pa;
      *(v4u*)(&Bs[row * LDA + col]) = pb;
    }
    __syncthreads();
#pragma unroll
    for (int kk = 0; kk < 64; kk += 32) {
      v16bf a0 = frag_a(As + (wm + 0) * LDA + kk, LDA, lane);
      v16bf a1 = frag_a(As + (wm + 16) * LDA + kk, LDA, lane);
      v16bf b0 = frag_b(Bs + (wn + 0) * LDA + kk, LDA, lane);
      v16bf b1 = frag_b(Bs + (wn + 16) * LDA + kk, LDA, lane);
      acc00 = WMMA_BF16(a0, b0, acc00);
      acc01 = WMMA_BF16(a0, b1, acc01);
      acc10 = WMMA_BF16(a1, b0, acc10);
      acc11 = WMMA_BF16(a1, b1, acc11);
    }
    __syncthreads();
  }

  // branch-free epilogue: layout difference folded into uniform multipliers
  unsigned short* dst = (sel == 0) ? qh : (sel == 1 ? kh : vT);
  const int lm = (sel == 2) ? 1 : 64;     // stride of sequence index l
  const int hm = (sel == 2) ? 1024 : 1;   // stride of head-dim index hd
  const int hs = lane >> 4, nl = lane & 15;
  v8f accs[2][2] = {{acc00, acc01}, {acc10, acc11}};
#pragma unroll
  for (int ti = 0; ti < 2; ++ti)
#pragma unroll
    for (int tj = 0; tj < 2; ++tj)
#pragma unroll
      for (int v = 0; v < 8; ++v) {
        const int r = m0 + wm + ti * 16 + hs * 8 + v;  // global row in [0,4096)
        const int c = n0 + wn + tj * 16 + nl;          // global col in [0,1024)
        const int b = r >> 10, l = r & 1023;
        const int h = c >> 6, hd = c & 63;
        dst[((size_t)(b * 16 + h) << 16) + l * lm + hd * hm] = f2bf(accs[ti][tj][v]);
      }
}

// ---------------------------------------------------------------------------
// Kernel 2: flash attention. One wave per 16 query rows, 64-key tiles,
// online softmax (f32), masking per reference (-1e6, valid_lens[bh*L+q]).
// ---------------------------------------------------------------------------
__global__ __launch_bounds__(128) void flash_attn_kernel(
    const unsigned short* __restrict__ qh, const unsigned short* __restrict__ kh,
    const unsigned short* __restrict__ vT, const int* __restrict__ vlens,
    unsigned short* __restrict__ attn_out) {
  constexpr int LDP = 72;  // conflict-free padded stride, 16B-aligned rows
  __shared__ __align__(16) unsigned short Plds[4 * 16 * LDP];

  const int bh = blockIdx.y;  // 0..63
  const int qt = blockIdx.x;  // 0..15
  const int t = threadIdx.x, lane = t & 31, w = t >> 5;
  const int hs = lane >> 4, nl = lane & 15;
  unsigned short* Pl = Plds + w * 16 * LDP;
  const int q0 = qt * 64 + w * 16;

  // Q A-fragments, loaded once (layout in qh matches WMMA A striping directly)
  const unsigned short* Qb = qh + ((size_t)bh * 1024 + q0) * 64;
  const v16bf qa0 = frag_a(Qb, 64, lane);       // hd 0..31
  const v16bf qa1 = frag_a(Qb + 32, 64, lane);  // hd 32..63

  v8f acc[4];
#pragma unroll
  for (int i = 0; i < 4; ++i) acc[i] = v8f{};

  float ms[8], ls[8];
  int vlen[8];
#pragma unroll
  for (int v = 0; v < 8; ++v) {
    ms[v] = -__builtin_inff();
    ls[v] = 0.f;
    vlen[v] = vlens[bh * 1024 + q0 + hs * 8 + v];
  }

  for (int kt = 0; kt < 1024; kt += 64) {
    // S = Q K^T (bf16 WMMA, f32 accum), 4 key sub-tiles of 16
    v8f s[4];
#pragma unroll
    for (int tj = 0; tj < 4; ++tj) {
      const unsigned short* Kb = kh + ((size_t)bh * 1024 + kt + tj * 16) * 64;
      v16bf kb0 = frag_b(Kb, 64, lane);
      v16bf kb1 = frag_b(Kb + 32, 64, lane);
      v8f c = {};
      c = WMMA_BF16(qa0, kb0, c);
      c = WMMA_BF16(qa1, kb1, c);
      s[tj] = c;
    }
    // online softmax per query row (rows live in 16-lane halves of the wave)
#pragma unroll
    for (int v = 0; v < 8; ++v) {
      float x = -__builtin_inff();
#pragma unroll
      for (int tj = 0; tj < 4; ++tj) {
        float e = s[tj][v] * 0.125f;  // 1/sqrt(64)
        const int col = kt + tj * 16 + nl;
        if (col >= vlen[v]) e = -1.0e6f;
        s[tj][v] = e;
        x = fmaxf(x, e);
      }
      x = fmaxf(x, __shfl_xor(x, 1, 32));
      x = fmaxf(x, __shfl_xor(x, 2, 32));
      x = fmaxf(x, __shfl_xor(x, 4, 32));
      x = fmaxf(x, __shfl_xor(x, 8, 32));
      const float mnew = fmaxf(ms[v], x);
      const float alpha = __expf(ms[v] - mnew);
      float rs = 0.f;
#pragma unroll
      for (int tj = 0; tj < 4; ++tj) {
        const float p = __expf(s[tj][v] - mnew);
        s[tj][v] = p;
        rs += p;
      }
      rs += __shfl_xor(rs, 1, 32);
      rs += __shfl_xor(rs, 2, 32);
      rs += __shfl_xor(rs, 4, 32);
      rs += __shfl_xor(rs, 8, 32);
      ls[v] = ls[v] * alpha + rs;
      ms[v] = mnew;
#pragma unroll
      for (int tn = 0; tn < 4; ++tn) acc[tn][v] *= alpha;
      // stage P (C-layout -> row-major LDS) for A-fragment reload
#pragma unroll
      for (int tj = 0; tj < 4; ++tj)
        Pl[(hs * 8 + v) * LDP + tj * 16 + nl] = f2bf(s[tj][v]);
    }
    // O += P @ V : P as A-frag from LDS, V as B-frag straight from vT (global)
#pragma unroll
    for (int kk = 0; kk < 2; ++kk) {
      v16bf pa = frag_a(Pl + kk * 32, LDP, lane);
#pragma unroll
      for (int tn = 0; tn < 4; ++tn) {
        const unsigned short* Vb =
            vT + ((size_t)bh * 64 + tn * 16) * 1024 + kt + kk * 32;
        v16bf vb = frag_b(Vb, 1024, lane);
        acc[tn] = WMMA_BF16(pa, vb, acc[tn]);
      }
    }
  }

  // normalize and store merged-head bf16 [B][L][D]
  const int b = bh >> 4, h = bh & 15;
#pragma unroll
  for (int tn = 0; tn < 4; ++tn)
#pragma unroll
    for (int v = 0; v < 8; ++v) {
      const float o = acc[tn][v] / ls[v];
      const int q = q0 + hs * 8 + v;
      const int col = h * 64 + tn * 16 + nl;
      attn_out[((size_t)b * 1024 + q) * 1024 + col] = f2bf(o);
    }
}

// ---------------------------------------------------------------------------
// Kernel 3: output projection. out[4096x1024] (f32) = attn_out(bf16) @ Wo^T.
// K-tile = 64, same structure as kernel 1.
// ---------------------------------------------------------------------------
__global__ __launch_bounds__(128) void out_proj_kernel(
    const unsigned short* __restrict__ A, const float* __restrict__ Wo,
    float* __restrict__ out) {
  constexpr int LDA = 72;
  __shared__ __align__(16) unsigned short As[64 * LDA];
  __shared__ __align__(16) unsigned short Bs[64 * LDA];

  const int m0 = blockIdx.y * 64, n0 = blockIdx.x * 64;
  const int t = threadIdx.x, lane = t & 31, w = t >> 5;
  const int wm = (w & 1) * 32, wn = (w >> 1) * 32;

  v8f acc00 = {}, acc01 = {}, acc10 = {}, acc11 = {};

  for (int kt = 0; kt < 1024; kt += 64) {
    if (kt + 64 < 1024) {
      __builtin_prefetch(A + (size_t)(m0 + t) * 1024 + (kt + 64), 0, 0);
      __builtin_prefetch(Wo + (size_t)(n0 + (t >> 1)) * 1024 + (kt + 64) + (t & 1) * 32, 0, 0);
    }
#pragma unroll
    for (int i = 0; i < 4; ++i) {  // A tile: 512 8-ushort chunks (bf16 direct)
      const int c = i * 128 + t;
      const int row = c >> 3;          // 8 chunks per 64-wide row
      const int col = (c & 7) * 8;
      v8u va = *(const v8u*)(A + (size_t)(m0 + row) * 1024 + kt + col);
      *(v8u*)(&As[row * LDA + col]) = va;
    }
#pragma unroll
    for (int i = 0; i < 8; ++i) {  // Wo tile: f32 -> bf16
      const int c = i * 128 + t;
      const int row = c >> 4;
      const int col = (c & 15) * 4;
      v4f wb = *(const v4f*)(Wo + (size_t)(n0 + row) * 1024 + kt + col);
      v4u pb = {f2bf(wb[0]), f2bf(wb[1]), f2bf(wb[2]), f2bf(wb[3])};
      *(v4u*)(&Bs[row * LDA + col]) = pb;
    }
    __syncthreads();
#pragma unroll
    for (int kk = 0; kk < 64; kk += 32) {
      v16bf a0 = frag_a(As + (wm + 0) * LDA + kk, LDA, lane);
      v16bf a1 = frag_a(As + (wm + 16) * LDA + kk, LDA, lane);
      v16bf b0 = frag_b(Bs + (wn + 0) * LDA + kk, LDA, lane);
      v16bf b1 = frag_b(Bs + (wn + 16) * LDA + kk, LDA, lane);
      acc00 = WMMA_BF16(a0, b0, acc00);
      acc01 = WMMA_BF16(a0, b1, acc01);
      acc10 = WMMA_BF16(a1, b0, acc10);
      acc11 = WMMA_BF16(a1, b1, acc11);
    }
    __syncthreads();
  }

  const int hs = lane >> 4, nl = lane & 15;
  v8f accs[2][2] = {{acc00, acc01}, {acc10, acc11}};
#pragma unroll
  for (int ti = 0; ti < 2; ++ti)
#pragma unroll
    for (int tj = 0; tj < 2; ++tj)
#pragma unroll
      for (int v = 0; v < 8; ++v) {
        const int r = m0 + wm + ti * 16 + hs * 8 + v;
        const int c = n0 + wn + tj * 16 + nl;
        out[(size_t)r * 1024 + c] = accs[ti][tj][v];
      }
}

// ---------------------------------------------------------------------------
extern "C" void kernel_launch(void* const* d_in, const int* in_sizes, int n_in,
                              void* d_out, int out_size, void* d_ws, size_t ws_size,
                              hipStream_t stream) {
  const float* Q = (const float*)d_in[0];
  const float* K = (const float*)d_in[1];
  const float* V = (const float*)d_in[2];
  const int* vl = (const int*)d_in[3];
  const float* Wq = (const float*)d_in[4];
  const float* Wk = (const float*)d_in[5];
  const float* Wv = (const float*)d_in[6];
  const float* Wo = (const float*)d_in[7];

  // workspace: 4 bf16 buffers of 64*1024*64 = 4M elements (8 MB) each = 32 MB
  unsigned short* qh = (unsigned short*)d_ws;
  unsigned short* kh = qh + (size_t)64 * 1024 * 64;
  unsigned short* vT = kh + (size_t)64 * 1024 * 64;
  unsigned short* ao = vT + (size_t)64 * 1024 * 64;

  qkv_proj_kernel<<<dim3(16, 64, 3), 128, 0, stream>>>(Q, K, V, Wq, Wk, Wv, qh, kh, vT);
  flash_attn_kernel<<<dim3(16, 64), 128, 0, stream>>>(qh, kh, vT, vl, ao);
  out_proj_kernel<<<dim3(16, 64), 128, 0, stream>>>(ao, Wo, (float*)d_out);
}
